// CRF_48266842472844
// MI455X (gfx1250) — matrix operand order repaired and checked by reading, not compiled, not run on hardware
//
#include <hip/hip_runtime.h>

typedef __attribute__((ext_vector_type(16))) _Float16 v16h;
typedef __attribute__((ext_vector_type(8)))  float    v8f;
typedef __attribute__((ext_vector_type(4)))  float    f4;

#define B_  256
#define T_  512
#define L_  64
#define RS  68        // LDS row stride in dwords: (4r+k)%64 distinct per row -> conflict-free
#define PF  12        // prefetch distance in timesteps

#define LOG2E 1.44269504088896340736f   // 1/ln(2): natural-domain emit -> log2 domain
#define LN2   0.69314718055994530942f   // final log2 -> natural log

// ---------------------------------------------------------------------------
// Kernel 2: CRF forward scan with LINEAR state + per-row power-of-2-ish scale.
// One wave32 handles a 16-batch tile for all T.
//   q      = S_{t-1} * exp2(emit_{t-1}*log2e)      (A-pattern, LDS + regs)
//   mx     = rowmax(q);  P = q * rcp(mx);  C += log2(mx)
//   S_t    = P @ E   (8x v_wmma_f32_16x16x32_f16)  -> stored raw to LDS
// Invariant: emit-deferred state gamma_t = S_t * 2^C_t.
// E = exp(trans) lives in registers (WMMA B-layout) for the whole scan.
// ---------------------------------------------------------------------------
__global__ __launch_bounds__(32) void crf_forward_kernel(
    const float* __restrict__ y_pred,   // (B,T,L)
    const float* __restrict__ trans,    // (L,L)
    const float* __restrict__ score,    // (B,) point+trans scores from kernel 1
    float* __restrict__ out)            // (B,)
{
    __shared__ __align__(16) float Sbuf[16 * RS]; // linear S_t per [row][state]

    const int lane  = threadIdx.x;     // 0..31
    const int m     = lane & 15;       // batch row within tile (A/B/D minor layouts)
    const int h     = lane >> 4;       // K-half select (A/B layouts)
    const int baseb = blockIdx.x * 16;
    const float* rowp = y_pred + (size_t)(baseb + m) * T_ * L_;

    // ---- Build E[i,j] = exp(trans[i,j]) in WMMA B-layout registers -------
    // B 32x16 f16: elem e of lane half h -> K = (e&7) + 8h + 16*(e>>3) (matches A)
    v16h E[4][2];
    #pragma unroll
    for (int nt = 0; nt < 4; ++nt)
        #pragma unroll
        for (int c = 0; c < 2; ++c)
            #pragma unroll
            for (int e = 0; e < 16; ++e) {
                const int i = 32*c + 8*h + 16*(e >> 3) + (e & 7);   // source state (K dim)
                const int j = 16*nt + m;         // dest state   (N dim)
                E[nt][c][e] = (_Float16)__builtin_amdgcn_exp2f(trans[i * L_ + j] * LOG2E);
            }

    // ---- init: S_0 = 1, C = 0  (state_0 = emit_0, applied at first step) -
    for (int idx = lane; idx < 16 * RS; idx += 32) Sbuf[idx] = 1.0f;
    float C = 0.0f;    // per-row accumulated log2 scale (lane l owns row l&15)
    __syncthreads();

    // ---- preload emit_0 in A-pattern: K = (e&7) + 8h + 16*(e>>3) + 32c ---
    f4 emv[8];
    #pragma unroll
    for (int c = 0; c < 2; ++c)
        #pragma unroll
        for (int g = 0; g < 2; ++g) {
            const int K0 = 8*h + 16*g + 32*c;
            emv[c*4 + g*2 + 0] = *(const f4*)(rowp + K0);
            emv[c*4 + g*2 + 1] = *(const f4*)(rowp + K0 + 4);
        }

    // ======================= serial scan over time ========================
    for (int t = 1; t < T_; ++t) {
        if (t + PF < T_)
            __builtin_prefetch(rowp + (size_t)(t + PF) * L_ + 32*h, 0, 1);

        // Phase A: q = S_{t-1} * exp2(emit*log2e). The exps depend only on
        // the global loads issued last iteration -> off the LDS-read chain.
        float vv[32];
        #pragma unroll
        for (int c = 0; c < 2; ++c)
            #pragma unroll
            for (int g = 0; g < 2; ++g) {
                const int K0 = 8*h + 16*g + 32*c;
                const f4 s0 = *(const f4*)&Sbuf[m*RS + K0];
                const f4 s1 = *(const f4*)&Sbuf[m*RS + K0 + 4];
                const f4 e0 = emv[c*4 + g*2 + 0];
                const f4 e1 = emv[c*4 + g*2 + 1];
                #pragma unroll
                for (int q = 0; q < 4; ++q) {
                    vv[c*16 + g*8 + q]     = s0[q] * __builtin_amdgcn_exp2f(e0[q] * LOG2E);
                    vv[c*16 + g*8 + 4 + q] = s1[q] * __builtin_amdgcn_exp2f(e1[q] * LOG2E);
                }
            }

        // issue next step's emit loads now (consumed next iteration / finale)
        {
            const float* p = rowp + (size_t)t * L_;
            #pragma unroll
            for (int c = 0; c < 2; ++c)
                #pragma unroll
                for (int g = 0; g < 2; ++g) {
                    const int K0 = 8*h + 16*g + 32*c;
                    emv[c*4 + g*2 + 0] = *(const f4*)(p + K0);
                    emv[c*4 + g*2 + 1] = *(const f4*)(p + K0 + 4);
                }
        }

        // Row max: tree over lane's 32 values, then swap with K-complement lane
        float tm[16];
        #pragma unroll
        for (int i = 0; i < 16; ++i) tm[i] = fmaxf(vv[i], vv[i + 16]);
        #pragma unroll
        for (int i = 0; i < 8;  ++i) tm[i] = fmaxf(tm[i], tm[i + 8]);
        #pragma unroll
        for (int i = 0; i < 4;  ++i) tm[i] = fmaxf(tm[i], tm[i + 4]);
        float mx = fmaxf(fmaxf(tm[0], tm[1]), fmaxf(tm[2], tm[3]));
        mx = fmaxf(mx, __shfl_xor(mx, 16, 32));

        // Normalize: P = q/mx in [0,1], perfect for the f16 A operand.
        const float inv = __builtin_amdgcn_rcpf(mx);
        C += __builtin_amdgcn_logf(mx);

        v16h a0, a1;
        #pragma unroll
        for (int e = 0; e < 16; ++e) {
            a0[e] = (_Float16)(vv[e]      * inv);
            a1[e] = (_Float16)(vv[16 + e] * inv);
        }

        __syncthreads();   // all reads of Sbuf done before rewrite (cheap: 1 wave)

        // S = P @ E via WMMA; store raw f32 accumulator to LDS (D-layout M=r+8h)
        #pragma unroll
        for (int nt = 0; nt < 4; ++nt) {
            v8f acc = {};
            acc = __builtin_amdgcn_wmma_f32_16x16x32_f16(
                      false, a0, false, E[nt][0], (short)0, acc, false, false);
            acc = __builtin_amdgcn_wmma_f32_16x16x32_f16(
                      false, a1, false, E[nt][1], (short)0, acc, false, false);
            #pragma unroll
            for (int r = 0; r < 8; ++r) {
                const int M = r + 8*h;
                Sbuf[M*RS + 16*nt + m] = acc[r];
            }
        }
        __syncthreads();
    }

    // ---- finale: log_norm[b] = ln2 * ( C + log2( sum_j S*exp2(emit2) ) ) -
    float s = 0.0f;
    #pragma unroll
    for (int c = 0; c < 2; ++c)
        #pragma unroll
        for (int g = 0; g < 2; ++g) {
            const int K0 = 8*h + 16*g + 32*c;
            const f4 s0 = *(const f4*)&Sbuf[m*RS + K0];
            const f4 s1 = *(const f4*)&Sbuf[m*RS + K0 + 4];
            const f4 e0 = emv[c*4 + g*2 + 0];
            const f4 e1 = emv[c*4 + g*2 + 1];
            #pragma unroll
            for (int q = 0; q < 4; ++q) {
                s += s0[q] * __builtin_amdgcn_exp2f(e0[q] * LOG2E);
                s += s1[q] * __builtin_amdgcn_exp2f(e1[q] * LOG2E);
            }
        }
    s += __shfl_xor(s, 16, 32);
    const float log_norm = (C + __builtin_amdgcn_logf(s)) * LN2;

    if (lane < 16)
        out[baseb + lane] = log_norm - score[baseb + lane];
}

// ---------------------------------------------------------------------------
// Kernel 1: point_score + trans_score per batch -> d_ws. Memory bound (~67MB).
// ---------------------------------------------------------------------------
__global__ __launch_bounds__(128) void crf_score_kernel(
    const float* __restrict__ y_true,   // (B,T,L) one-hot
    const float* __restrict__ y_pred,   // (B,T,L)
    const float* __restrict__ trans,    // (L,L)
    float* __restrict__ score)          // (B,)
{
    __shared__ int   labs[T_];
    __shared__ float red[128];
    const int b = blockIdx.x, tid = threadIdx.x;

    float acc = 0.0f;
    for (int t = tid; t < T_; t += 128) {
        const float* yt = y_true + ((size_t)b * T_ + t) * L_;
        const float* yp = y_pred + ((size_t)b * T_ + t) * L_;
        float labf = 0.0f, pt = 0.0f;
        #pragma unroll 4
        for (int k = 0; k < L_; k += 4) {
            const f4 a = *(const f4*)(yt + k);
            const f4 p = *(const f4*)(yp + k);
            labf += a[0]*(float)k + a[1]*(float)(k+1) + a[2]*(float)(k+2) + a[3]*(float)(k+3);
            pt   += a[0]*p[0] + a[1]*p[1] + a[2]*p[2] + a[3]*p[3];
        }
        labs[t] = (int)(labf + 0.5f);
        acc += pt;
    }
    __syncthreads();
    for (int t = tid; t < T_ - 1; t += 128)
        acc += trans[labs[t] * L_ + labs[t + 1]];

    red[tid] = acc;
    __syncthreads();
    for (int s = 64; s > 0; s >>= 1) {
        if (tid < s) red[tid] += red[tid + s];
        __syncthreads();
    }
    if (tid == 0) score[b] = red[0];
}

// ---------------------------------------------------------------------------
extern "C" void kernel_launch(void* const* d_in, const int* in_sizes, int n_in,
                              void* d_out, int out_size, void* d_ws, size_t ws_size,
                              hipStream_t stream)
{
    const float* y_true = (const float*)d_in[0];
    const float* y_pred = (const float*)d_in[1];
    const float* trans  = (const float*)d_in[2];
    float* out   = (float*)d_out;
    float* score = (float*)d_ws;          // B floats of scratch

    crf_score_kernel<<<B_, 128, 0, stream>>>(y_true, y_pred, trans, score);
    crf_forward_kernel<<<B_ / 16, 32, 0, stream>>>(y_pred, trans, score, out);
}
